// LevelSetLayer2D_42004780155575
// MI455X (gfx1250) — matrix operand order repaired and checked by reading, not compiled, not run on hardware
//
#include <hip/hip_runtime.h>
#include <stdint.h>

typedef __attribute__((ext_vector_type(16))) _Float16 v16h;
typedef __attribute__((ext_vector_type(8)))  float    v8f;

#define NT   256
#define MAXN 8192
#define COLW 256   // max words per bit-packed column (n <= 8192)

#if __has_builtin(__builtin_amdgcn_global_load_async_to_lds_b32) && \
    __has_builtin(__builtin_amdgcn_s_wait_asynccnt)
#define HAVE_ASYNC_LDS 1
typedef __attribute__((address_space(1))) int async_g_i32;
typedef __attribute__((address_space(3))) int async_l_i32;
#else
#define HAVE_ASYNC_LDS 0
#endif

// ---------------------------------------------------------------------------
// helpers
// ---------------------------------------------------------------------------
__device__ __forceinline__ unsigned sortable_bits(float f) {
  unsigned u = __float_as_uint(f);
  return u ^ ((u >> 31) ? 0xFFFFFFFFu : 0x80000000u);
}

__device__ __forceinline__ unsigned gload_agent(const unsigned* p) {
  return __hip_atomic_load(p, __ATOMIC_RELAXED, __HIP_MEMORY_SCOPE_AGENT);
}
__device__ __forceinline__ void gstore_agent(unsigned* p, unsigned v) {
  __hip_atomic_store(p, v, __ATOMIC_RELAXED, __HIP_MEMORY_SCOPE_AGENT);
}

// ---------------------------------------------------------------------------
// zero-fill (harness poisons d_out / d_ws with 0xAA)
// ---------------------------------------------------------------------------
__global__ void k_zero(unsigned* p, long long count) {
  long long i = (long long)blockIdx.x * blockDim.x + threadIdx.x;
  long long stride = (long long)gridDim.x * blockDim.x;
  for (; i < count; i += stride) p[i] = 0u;
}

// ---------------------------------------------------------------------------
// fs[k] = max over vertices; 64-bit lexsort key (fs, dims, index)
// ---------------------------------------------------------------------------
__global__ void k_init(const float* fv, const int* verts, const int* dims, int n,
                       float* fs, unsigned long long* keys) {
  int k = blockIdx.x * blockDim.x + threadIdx.x;
  if (k >= n) return;
  float a = fv[verts[3 * k + 0]];
  float b = fv[verts[3 * k + 1]];
  float c = fv[verts[3 * k + 2]];
  float m = fmaxf(a, fmaxf(b, c));
  fs[k] = m;
  unsigned su = sortable_bits(m);
  keys[k] = ((unsigned long long)su << 32) |
            ((unsigned long long)(unsigned)dims[k] << 16) |
            (unsigned)k;  // n < 65536, dims < 4
}

// ---------------------------------------------------------------------------
// rank by counting (keys are distinct -> permutation); scatter sorted arrays
// ---------------------------------------------------------------------------
__global__ void k_rank(const unsigned long long* keys, int n,
                       const float* fs, const int* dims,
                       int* rank, float* fs_s, int* dims_s) {
  __shared__ unsigned long long tile[NT];
  int k = blockIdx.x * blockDim.x + threadIdx.x;
  unsigned long long my = (k < n) ? keys[k] : ~0ull;
  int r = 0;
  for (int base = 0; base < n; base += NT) {
    int idx = base + threadIdx.x;
    tile[threadIdx.x] = (idx < n) ? keys[idx] : ~0ull;
    if (base + NT + (int)threadIdx.x < n)
      __builtin_prefetch(&keys[base + NT + threadIdx.x], 0, 0);  // global_prefetch_b8
    __syncthreads();
    int lim = min(NT, n - base);
    for (int t = 0; t < lim; ++t) r += (tile[t] < my) ? 1 : 0;
    __syncthreads();
  }
  if (k < n) {
    rank[k]   = r;
    fs_s[r]   = fs[k];
    dims_s[r] = dims[k];
  }
}

// ---------------------------------------------------------------------------
// build bit-packed Z2 boundary matrix in sorted order
// ---------------------------------------------------------------------------
__global__ void k_build(const int* faces, const int* rank, int n, int W,
                        unsigned* colbits) {
  int k = blockIdx.x * blockDim.x + threadIdx.x;
  if (k >= n) return;
  int j = rank[k];
  size_t base = (size_t)j * W;
  for (int e = 0; e < 3; ++e) {
    int fc = faces[3 * k + e];
    if (fc >= 0) {
      int row = rank[fc];
      colbits[base + (row >> 5)] |= (1u << (row & 31));  // column private to thread
    }
  }
}

// ---------------------------------------------------------------------------
// standard persistence reduction: one workgroup, column in LDS,
// parallel low() reduce + parallel XOR against pivot columns.
// Initial column load has no intra-kernel RAW hazard (each column is written
// back only at its own iteration) -> use CDNA5 async global->LDS DMA.
// ---------------------------------------------------------------------------
__global__ void k_reduce(unsigned* colbits, int n, int W, int* pivot, int* lowcol) {
  __shared__ int      s_pivot[MAXN];
  __shared__ unsigned s_col[COLW];
  __shared__ int      s_low;
  int tid = threadIdx.x, nt = blockDim.x;

  for (int i = tid; i < n; i += nt) s_pivot[i] = -1;
  __syncthreads();

  for (int j = 0; j < n; ++j) {
    unsigned* gcol = colbits + (size_t)j * W;
#if HAVE_ASYNC_LDS
    for (int w = tid; w < W; w += nt) {
      __builtin_amdgcn_global_load_async_to_lds_b32(
          (async_g_i32*)(uintptr_t)(&gcol[w]),
          (async_l_i32*)(unsigned)(uintptr_t)(&s_col[w]),
          0, 0);
    }
    __builtin_amdgcn_s_wait_asynccnt(0);
#else
    for (int w = tid; w < W; w += nt) s_col[w] = gload_agent(&gcol[w]);
#endif
    __syncthreads();
    while (true) {
      if (tid == 0) s_low = -1;
      __syncthreads();
      for (int w = tid; w < W; w += nt) {
        unsigned x = s_col[w];
        if (x) atomicMax(&s_low, w * 32 + (31 - __builtin_clz(x)));
      }
      __syncthreads();
      int low = s_low;
      if (low < 0) {                       // column reduced to zero -> birth
        if (tid == 0) lowcol[j] = -1;
        break;
      }
      int p = s_pivot[low];
      if (p < 0) {                         // claim pivot
        if (tid == 0) { s_pivot[low] = j; lowcol[j] = low; }
        break;
      }
      unsigned* pcol = colbits + (size_t)p * W;   // intra-kernel RAW: coherent loads
      for (int w = tid; w < W; w += nt) s_col[w] ^= gload_agent(&pcol[w]);
      __syncthreads();
    }
    __syncthreads();
    for (int w = tid; w < W; w += nt) gstore_agent(&gcol[w], s_col[w]);
    __syncthreads();
  }
  for (int i = tid; i < n; i += nt) pivot[i] = s_pivot[i];
}

// ---------------------------------------------------------------------------
// WMMA 256-element inclusive prefix sum (exact: 0/1 masks, counts <= 256)
// X is 16x16 (row-major over the chunk). Y = X*U (U = upper-tri ones) gives
// row-wise prefixes; P = Lstrict*T + Y (T broadcasts row totals) completes it.
// ---------------------------------------------------------------------------
__device__ __forceinline__ void wave_prefix256(const _Float16* hv, float* tt,
                                               int lane, int* s_pref, int* s_ctot,
                                               int c) {
  int M = lane & 15;
  int half = lane >> 4;
  int N = M;

  v16h a, b;
  // A fragment (16-bit A 16x32 layout): lane L holds row M=L&15;
  // VGPR v in 0..3 packs K = 2v+8*half, +1; VGPRs 4..7 (K>=16) zero-padded.
#pragma unroll
  for (int v = 0; v < 4; ++v) {
    int k0 = 2 * v + 8 * half;
    a[2 * v]     = hv[M * 16 + k0];
    a[2 * v + 1] = hv[M * 16 + k0 + 1];
  }
#pragma unroll
  for (int v = 4; v < 8; ++v) { a[2 * v] = (_Float16)0.f; a[2 * v + 1] = (_Float16)0.f; }
  // B fragment (32x16, same K order as A): lane holds col N; VGPR v in 0..3 packs
  // K = 2v+8*half, +1. U[k][n] = 1 iff k <= n; VGPRs 4..7 (K>=16) are zero.
#pragma unroll
  for (int v = 0; v < 8; ++v) {
    int k0 = 2 * v + 8 * half;
    b[2 * v]     = (_Float16)((v < 4 && k0     <= N) ? 1.f : 0.f);
    b[2 * v + 1] = (_Float16)((v < 4 && k0 + 1 <= N) ? 1.f : 0.f);
  }
  v8f acc = {};
  acc = __builtin_amdgcn_wmma_f32_16x16x32_f16(false, a, false, b, (short)0, acc,
                                               false, false);
  // row totals t[m] = Y[m][15], held by lanes with N==15 (VGPR g -> M=g+8*half)
  if (N == 15) {
#pragma unroll
    for (int g = 0; g < 8; ++g) tt[g + 8 * half] = acc[g];
  }
  __builtin_amdgcn_wave_barrier();

  v16h a2, b2;
  // A2 = strict lower triangular ones
#pragma unroll
  for (int v = 0; v < 4; ++v) {
    int k0 = 2 * v + 8 * half;
    a2[2 * v]     = (_Float16)((k0     < M) ? 1.f : 0.f);
    a2[2 * v + 1] = (_Float16)((k0 + 1 < M) ? 1.f : 0.f);
  }
#pragma unroll
  for (int v = 4; v < 8; ++v) { a2[2 * v] = (_Float16)0.f; a2[2 * v + 1] = (_Float16)0.f; }
  // B2: T[k][n] = t[k] for k < 16 (VGPR v in 0..3 packs K = 2v+8*half, +1),
  // zero for VGPRs 4..7 (K-rows 16..31).
#pragma unroll
  for (int v = 0; v < 4; ++v) {
    b2[2 * v]     = (_Float16)tt[2 * v + 8 * half];
    b2[2 * v + 1] = (_Float16)tt[2 * v + 8 * half + 1];
  }
#pragma unroll
  for (int v = 4; v < 8; ++v) { b2[2 * v] = (_Float16)0.f; b2[2 * v + 1] = (_Float16)0.f; }
  acc = __builtin_amdgcn_wmma_f32_16x16x32_f16(false, a2, false, b2, (short)0, acc,
                                               false, false);
  // scatter inclusive prefixes: VGPR g -> (M=g+8*half, N)
#pragma unroll
  for (int g = 0; g < 8; ++g) {
    int Mi = g + 8 * half;
    s_pref[c * 256 + Mi * 16 + N] = (int)(acc[g] + 0.5f);
  }
  if (lane == 31) s_ctot[c] = (int)(acc[7] + 0.5f);  // P[15][15] = chunk total
}

// ---------------------------------------------------------------------------
// pack dgm0 / dgm1: masks -> WMMA chunk prefix sums -> chunk offsets -> scatter
// ---------------------------------------------------------------------------
__global__ void k_pack(const float* fs_s, const int* dims_s, const int* lowcol,
                       const int* pivot, int n, int V, int cap1, float* out) {
  __shared__ int      s_pref[MAXN];
  __shared__ _Float16 hv[8][256];
  __shared__ __align__(16) float tt[8][16];
  __shared__ int      s_ctot[64];
  __shared__ int      s_choff[64];

  __builtin_amdgcn_s_wait_tensorcnt(0);

  int tid  = threadIdx.x;
  int wave = tid >> 5;
  int lane = tid & 31;
  int C = (n + 255) >> 8;

  for (int sel = 0; sel < 2; ++sel) {
    __syncthreads();
    for (int c = wave; c < C; c += 8) {   // wave-uniform trip count
      for (int r = 0; r < 8; ++r) {
        int t = lane * 8 + r;
        int i = c * 256 + t;
        float mv = 0.f;
        if (i < n && lowcol[i] == -1 && dims_s[i] == sel) mv = 1.f;
        hv[wave][t] = (_Float16)mv;
      }
      __builtin_amdgcn_wave_barrier();
      wave_prefix256(&hv[wave][0], &tt[wave][0], lane, s_pref, s_ctot, c);
    }
    __syncthreads();
    if (tid == 0) {
      int run = 0;
      for (int c = 0; c < C; ++c) { s_choff[c] = run; run += s_ctot[c]; }
    }
    __syncthreads();
    int cap = sel ? cap1 : V;
    float* base = sel ? (out + 2 * (size_t)V) : out;
    for (int i = tid; i < n; i += blockDim.x) {
      if (lowcol[i] == -1 && dims_s[i] == sel) {
        int pos = s_choff[i >> 8] + s_pref[i] - 1;
        if (pos >= 0 && pos < cap) {
          base[2 * pos + 0] = fs_s[i];
          int pv = pivot[i];
          base[2 * pos + 1] = (pv >= 0) ? fs_s[pv] : __builtin_inff();
        }
      }
    }
    __syncthreads();
  }
}

// ---------------------------------------------------------------------------
// host launcher
// ---------------------------------------------------------------------------
extern "C" void kernel_launch(void* const* d_in, const int* in_sizes, int n_in,
                              void* d_out, int out_size, void* d_ws, size_t ws_size,
                              hipStream_t stream) {
  const float* f     = (const float*)d_in[0];
  const int*   verts = (const int*)d_in[1];
  const int*   faces = (const int*)d_in[2];
  const int*   dims  = (const int*)d_in[3];
  (void)n_in; (void)ws_size;

  int V    = in_sizes[0];
  int n    = in_sizes[1] / 3;
  int cap1 = out_size / 2 - V;
  int W    = (n + 31) / 32;
  int NP   = ((n + 255) / 256) * 256;

  char* ws = (char*)d_ws;
  size_t off = 0;
  auto alloc = [&](size_t bytes) {
    void* p = ws + off;
    off += (bytes + 255) & ~(size_t)255;
    return p;
  };
  unsigned long long* keys = (unsigned long long*)alloc((size_t)NP * 8);
  float*    fs      = (float*)alloc((size_t)NP * 4);
  int*      rank    = (int*)alloc((size_t)NP * 4);
  float*    fs_s    = (float*)alloc((size_t)NP * 4);
  int*      dims_s  = (int*)alloc((size_t)NP * 4);
  int*      lowcol  = (int*)alloc((size_t)NP * 4);
  int*      pivot   = (int*)alloc((size_t)NP * 4);
  unsigned* colbits = (unsigned*)alloc((size_t)n * W * 4);

  int nb = (n + NT - 1) / NT;
  k_zero<<<256, NT, 0, stream>>>(colbits, (long long)n * W);
  k_zero<<<64, NT, 0, stream>>>((unsigned*)d_out, (long long)out_size);
  k_init<<<nb, NT, 0, stream>>>(f, verts, dims, n, fs, keys);
  k_rank<<<nb, NT, 0, stream>>>(keys, n, fs, dims, rank, fs_s, dims_s);
  k_build<<<nb, NT, 0, stream>>>(faces, rank, n, W, colbits);
  k_reduce<<<1, NT, 0, stream>>>(colbits, n, W, pivot, lowcol);
  k_pack<<<1, NT, 0, stream>>>(fs_s, dims_s, lowcol, pivot, n, V, cap1, (float*)d_out);
}
